// Self_Attention_Layer_5463198400761
// MI455X (gfx1250) — matrix operand order repaired and checked by reading, not compiled, not run on hardware
//
#include <hip/hip_runtime.h>
#include <stdint.h>
#include <stddef.h>

// ---------------- types ----------------
typedef __bf16 bf16_t;
typedef __attribute__((ext_vector_type(16))) __bf16 v16bf;
typedef __attribute__((ext_vector_type(8)))  __bf16 v8bf;
typedef __attribute__((ext_vector_type(4)))  __bf16 v4bf;
typedef __attribute__((ext_vector_type(8)))  float   v8f;

#define B_  16
#define S_  1024
#define D_  768
#define H_  12
#define HD_ 64
#define M_  (B_ * S_)        // 16384 rows of z

// ---------------- CDNA5 inline helpers ----------------
__device__ __forceinline__ void async_ld_b128(uint32_t lds_off, const bf16_t* gptr) {
  // GLOBAL_LOAD_ASYNC_TO_LDS_B128 (GV mode): copies 16 B from global vaddr to LDS at vdst, ASYNCcnt-tracked
  asm volatile("global_load_async_to_lds_b128 %0, %1, off"
               :: "v"(lds_off), "v"((uint64_t)(uintptr_t)gptr) : "memory");
}
__device__ __forceinline__ void wait_async0() {
  asm volatile("s_wait_asynccnt 0x0" ::: "memory");
}
__device__ __forceinline__ void wait_ds0() {
  asm volatile("s_wait_dscnt 0x0" ::: "memory");
}
__device__ __forceinline__ uint32_t lds_addr(const void* p) {
  return (uint32_t)(uintptr_t)p;     // low 32 bits of the shared pointer are the LDS offset
}

__device__ __forceinline__ v8f wmma_bf16(v16bf a, v16bf b, v8f c) {
  return __builtin_amdgcn_wmma_f32_16x16x32_bf16(false, a, false, b, (short)0, c,
                                                 false, false);
}

// A-fragment (16x32, M x K), row-major source, leading dim `ld` (elements).
// Lane l owns row (l&15); elems e -> K = e + 16*(l>>4).
__device__ __forceinline__ v16bf load_a_frag(const bf16_t* base, int ld, int lane) {
  const int row = lane & 15, grp = lane >> 4;
  const bf16_t* p = base + (size_t)row * ld + 16 * grp;
  return *(const v16bf*)(p);
}

// B-fragment (32x16, K x N) from a matrix row-major ALONG K: B[k][n] = Mrow[n][k].
// base points at (n0, k0). Lane l owns column (l&15); elems e -> K = e + 16*(l>>4).
__device__ __forceinline__ v16bf load_b_frag(const bf16_t* base, int ld, int lane) {
  const int n = lane & 15, grp = lane >> 4;
  return *(const v16bf*)(base + (size_t)n * ld + 16 * grp);
}

// ---------------- fp32 -> bf16 convert ----------------
__global__ __launch_bounds__(256) void cvt_f32_bf16(const float4* __restrict__ in,
                                                    bf16_t* __restrict__ out, int n4) {
  int i = blockIdx.x * blockDim.x + threadIdx.x;
  if (i >= n4) return;
  float4 v = in[i];
  v4bf o;
  o[0] = (bf16_t)v.x; o[1] = (bf16_t)v.y; o[2] = (bf16_t)v.z; o[3] = (bf16_t)v.w;
  *(v4bf*)(out + 4 * (size_t)i) = o;
}

// ---------------- GEMM mainloop (shared by qkv / out projections) ----------------
// Block = 8 waves; each wave owns 32 rows x 64 cols (2 A-frags, 8 accumulators,
// 8 WMMA per K-step). All waves share one 64-col W tile, staged in LDS with
// double-buffered global_load_async_to_lds_b128; one wait+barrier per K-step.
// All fragment loads of a K-step are issued before the WMMA chain.
__device__ __forceinline__ void gemm_mainloop(const bf16_t* __restrict__ A,
                                              const bf16_t* __restrict__ W,
                                              bf16_t (*wstage)[64 * 32],
                                              int m0, int n0, int tid, int lane,
                                              v8f acc[2][4]) {
  auto stage_w = [&](int k0, int buf) {
    const int row = tid >> 2, c = (tid & 3) * 8;
    async_ld_b128(lds_addr(&wstage[buf][row * 32 + c]),
                  W + (size_t)(n0 + row) * D_ + k0 + c);
  };

  stage_w(0, 0);
  wait_async0();
  __syncthreads();

  const bf16_t* arow0 = A + (size_t)m0 * D_;
  const bf16_t* arow1 = A + (size_t)(m0 + 16) * D_;

  int buf = 0;
  for (int k0 = 0; k0 < D_; k0 += 32) {
    if (k0 + 32 < D_) {
      stage_w(k0 + 32, buf ^ 1);                        // overlap with compute
      __builtin_prefetch(arow0 + k0 + 32, 0, 1);        // global_prefetch_b8
      __builtin_prefetch(arow1 + k0 + 32, 0, 1);
    }
    // issue ALL fragment loads for this K-step, then the WMMA chain
    v16bf a0 = load_a_frag(arow0 + k0, D_, lane);
    v16bf a1 = load_a_frag(arow1 + k0, D_, lane);
    v16bf b0 = load_b_frag(&wstage[buf][0 * 512], 32, lane);
    v16bf b1 = load_b_frag(&wstage[buf][1 * 512], 32, lane);
    v16bf b2 = load_b_frag(&wstage[buf][2 * 512], 32, lane);
    v16bf b3 = load_b_frag(&wstage[buf][3 * 512], 32, lane);
    acc[0][0] = wmma_bf16(a0, b0, acc[0][0]);
    acc[0][1] = wmma_bf16(a0, b1, acc[0][1]);
    acc[0][2] = wmma_bf16(a0, b2, acc[0][2]);
    acc[0][3] = wmma_bf16(a0, b3, acc[0][3]);
    acc[1][0] = wmma_bf16(a1, b0, acc[1][0]);
    acc[1][1] = wmma_bf16(a1, b1, acc[1][1]);
    acc[1][2] = wmma_bf16(a1, b2, acc[1][2]);
    acc[1][3] = wmma_bf16(a1, b3, acc[1][3]);
    wait_async0();
    __syncthreads();
    buf ^= 1;
  }
}

// ---------------- GEMM: proj = z @ W^T, scatter into head layouts ----------------
// transpose_out==0 -> [B,H,S,HD] (Q,K);  ==1 -> [B,H,HD,S] (V transposed).
__global__ __launch_bounds__(256) void gemm_qkv(const bf16_t* __restrict__ A,
                                                const bf16_t* __restrict__ W,
                                                bf16_t* __restrict__ out,
                                                int transpose_out) {
  __shared__ __align__(32) bf16_t wstage[2][64 * 32];   // 8 KB

  const int tid  = threadIdx.x;
  const int lane = tid & 31;
  const int wave = tid >> 5;
  const int mblk = blockIdx.x / (D_ / 64);              // 64 row-blocks of 256
  const int nt   = blockIdx.x % (D_ / 64);
  const int n0   = nt * 64;
  const int m0   = mblk * 256 + wave * 32;

  v8f zero = {};
  v8f acc[2][4] = {{zero, zero, zero, zero}, {zero, zero, zero, zero}};
  gemm_mainloop(A, W, wstage, m0, n0, tid, lane, acc);

  const int grp = lane >> 4, cl = lane & 15;
#pragma unroll
  for (int w2 = 0; w2 < 2; ++w2) {
#pragma unroll
    for (int j = 0; j < 4; ++j) {
      const int n = n0 + 16 * j + cl;
      const int h = n >> 6, d = n & 63;
#pragma unroll
      for (int r = 0; r < 8; ++r) {
        const int m = m0 + 16 * w2 + r + 8 * grp;
        const int b = m >> 10, s = m & 1023;
        size_t idx = transpose_out
                   ? ((size_t)((b * H_ + h) * HD_ + d) * S_ + s)
                   : ((size_t)((b * H_ + h) * S_ + s) * HD_ + d);
        out[idx] = (bf16_t)acc[w2][j][r];
      }
    }
  }
}

// ---------------- GEMM: out = ctx @ W_o^T + b_o (fp32 out) ----------------
__global__ __launch_bounds__(256) void gemm_out(const bf16_t* __restrict__ A,
                                                const bf16_t* __restrict__ W,
                                                const float* __restrict__ bias,
                                                float* __restrict__ out) {
  __shared__ __align__(32) bf16_t wstage[2][64 * 32];

  const int tid  = threadIdx.x;
  const int lane = tid & 31;
  const int wave = tid >> 5;
  const int mblk = blockIdx.x / (D_ / 64);
  const int nt   = blockIdx.x % (D_ / 64);
  const int n0   = nt * 64;
  const int m0   = mblk * 256 + wave * 32;

  v8f zero = {};
  v8f acc[2][4] = {{zero, zero, zero, zero}, {zero, zero, zero, zero}};
  gemm_mainloop(A, W, wstage, m0, n0, tid, lane, acc);

  const int grp = lane >> 4, cl = lane & 15;
#pragma unroll
  for (int w2 = 0; w2 < 2; ++w2) {
#pragma unroll
    for (int j = 0; j < 4; ++j) {
      const int n = n0 + 16 * j + cl;
      const float bo = bias[n];
#pragma unroll
      for (int r = 0; r < 8; ++r) {
        const int m = m0 + 16 * w2 + r + 8 * grp;
        out[(size_t)m * D_ + n] = acc[w2][j][r] + bo;
      }
    }
  }
}

// ---------------- flash attention ----------------
// block = 4 waves, one (b,h), 64 query rows (16 per wave). K chunk (32x64) and
// V^T chunk (64x32) are shared by the 4 waves: staged via async-LDS double
// buffering, one barrier per 32-key step.
__global__ __launch_bounds__(128) void flash_attn(const bf16_t* __restrict__ Q,
                                                  const bf16_t* __restrict__ K,
                                                  const bf16_t* __restrict__ VT,
                                                  bf16_t* __restrict__ ctx) {
  __shared__ __align__(32) bf16_t kstage[2][32 * 64];   // 8 KB  [key][d]
  __shared__ __align__(32) bf16_t vstage[2][64 * 32];   // 8 KB  [d][key]
  __shared__ __align__(32) bf16_t lds_p[4][16 * 32];    // 4 KB  per-wave P bounce

  const int tid  = threadIdx.x;
  const int lane = tid & 31;
  const int wave = tid >> 5;
  const int grp = lane >> 4, cl = lane & 15;
  const int bh   = blockIdx.x >> 4;                     // b*H + h
  const int qblk = blockIdx.x & 15;
  const int q0   = qblk * 64 + wave * 16;

  const bf16_t* Qh = Q  + (size_t)bh * S_ * HD_;
  const bf16_t* Kh = K  + (size_t)bh * S_ * HD_;
  const bf16_t* Vh = VT + (size_t)bh * HD_ * S_;        // [HD_ x S_] row-major

  auto stage_kv = [&](int kb, int buf) {
    const bf16_t* kg = Kh + (size_t)kb * HD_;
#pragma unroll
    for (int t = tid; t < 256; t += 128)
      async_ld_b128(lds_addr(&kstage[buf][t * 8]), kg + t * 8);
#pragma unroll
    for (int t = tid; t < 256; t += 128) {
      const int row = t >> 2, c = (t & 3) * 8;
      async_ld_b128(lds_addr(&vstage[buf][row * 32 + c]),
                    Vh + (size_t)row * S_ + kb + c);
    }
  };

  stage_kv(0, 0);

  // Q tile 16x64 -> two A-fragments (loaded once, overlaps the first copy)
  v16bf aq0 = load_a_frag(Qh + (size_t)q0 * HD_ +  0, HD_, lane);
  v16bf aq1 = load_a_frag(Qh + (size_t)q0 * HD_ + 32, HD_, lane);

  wait_async0();
  __syncthreads();

  v8f zero = {};
  v8f acc[4] = {zero, zero, zero, zero};                // ctx, d-tiles 0,16,32,48
  float m_run[8], l_run[8];
#pragma unroll
  for (int r = 0; r < 8; ++r) { m_run[r] = -1e30f; l_run[r] = 0.f; }

  const float scale = 0.125f;                           // 1/sqrt(64)
  bf16_t* myp = &lds_p[wave][0];

  int buf = 0;
  for (int kb = 0; kb < S_; kb += 32) {
    if (kb + 32 < S_) stage_kv(kb + 32, buf ^ 1);       // overlap with compute

    // ---- scores: issue all 4 K-fragment loads, then the WMMA chain ----
    v16bf bk00 = load_b_frag(&kstage[buf][ 0 * 64 +  0], 64, lane);
    v16bf bk01 = load_b_frag(&kstage[buf][ 0 * 64 + 32], 64, lane);
    v16bf bk10 = load_b_frag(&kstage[buf][16 * 64 +  0], 64, lane);
    v16bf bk11 = load_b_frag(&kstage[buf][16 * 64 + 32], 64, lane);
    v8f sc[2];
    sc[0] = wmma_bf16(aq0, bk00, zero);
    sc[1] = wmma_bf16(aq0, bk10, zero);
    sc[0] = wmma_bf16(aq1, bk01, sc[0]);
    sc[1] = wmma_bf16(aq1, bk11, sc[1]);

    // ---- online softmax (rows live across the 16-lane group) ----
#pragma unroll
    for (int r = 0; r < 8; ++r) {
      float a0 = sc[0][r] * scale;
      float a1 = sc[1][r] * scale;
      float mx = fmaxf(a0, a1);
      mx = fmaxf(mx, __shfl_xor(mx, 1, 32));
      mx = fmaxf(mx, __shfl_xor(mx, 2, 32));
      mx = fmaxf(mx, __shfl_xor(mx, 4, 32));
      mx = fmaxf(mx, __shfl_xor(mx, 8, 32));
      float mnew  = fmaxf(m_run[r], mx);
      float alpha = __expf(m_run[r] - mnew);
      float p0 = __expf(a0 - mnew);
      float p1 = __expf(a1 - mnew);
      float rs = p0 + p1;
      rs += __shfl_xor(rs, 1, 32);
      rs += __shfl_xor(rs, 2, 32);
      rs += __shfl_xor(rs, 4, 32);
      rs += __shfl_xor(rs, 8, 32);
      l_run[r] = l_run[r] * alpha + rs;
      m_run[r] = mnew;
      acc[0][r] *= alpha; acc[1][r] *= alpha;
      acc[2][r] *= alpha; acc[3][r] *= alpha;
      // stage P (bf16) for the D-frag -> A-frag transpose
      myp[(r + 8 * grp) * 32 + cl]      = (bf16_t)p0;
      myp[(r + 8 * grp) * 32 + cl + 16] = (bf16_t)p1;
    }
    wait_ds0();                                          // DS in-order per wave
    v16bf ap = load_a_frag(myp, 32, lane);

    // ---- ctx += P(16x32) @ V(32x64): all 4 V-fragments, then WMMA chain ----
    v16bf bv0 = load_b_frag(&vstage[buf][ 0 * 32], 32, lane);
    v16bf bv1 = load_b_frag(&vstage[buf][16 * 32], 32, lane);
    v16bf bv2 = load_b_frag(&vstage[buf][32 * 32], 32, lane);
    v16bf bv3 = load_b_frag(&vstage[buf][48 * 32], 32, lane);
    acc[0] = wmma_bf16(ap, bv0, acc[0]);
    acc[1] = wmma_bf16(ap, bv1, acc[1]);
    acc[2] = wmma_bf16(ap, bv2, acc[2]);
    acc[3] = wmma_bf16(ap, bv3, acc[3]);

    wait_async0();
    __syncthreads();
    buf ^= 1;
  }

  // ---- epilogue: normalize, scatter into ctx [B*S, D] bf16 ----
  const int b = bh / H_, h = bh % H_;
#pragma unroll
  for (int r = 0; r < 8; ++r) {
    float inv = 1.0f / l_run[r];
    const int srow = q0 + r + 8 * grp;
    size_t rowbase = (size_t)(b * S_ + srow) * D_ + h * HD_;
#pragma unroll
    for (int j = 0; j < 4; ++j)
      ctx[rowbase + 16 * j + cl] = (bf16_t)(acc[j][r] * inv);
  }
}

// ---------------- host launcher ----------------
extern "C" void kernel_launch(void* const* d_in, const int* in_sizes, int n_in,
                              void* d_out, int out_size, void* d_ws, size_t ws_size,
                              hipStream_t stream) {
  const float* z   = (const float*)d_in[0];
  const float* W_q = (const float*)d_in[1];
  const float* W_k = (const float*)d_in[2];
  const float* W_v = (const float*)d_in[3];
  const float* W_o = (const float*)d_in[4];
  const float* b_o = (const float*)d_in[5];
  float* out = (float*)d_out;

  char* ws = (char*)d_ws;
  size_t off = 0;
  auto carve = [&](size_t bytes) -> char* {
    char* p = ws + off;
    off += (bytes + 255) & ~(size_t)255;
    return p;
  };
  const size_t wbytes = (size_t)D_ * D_ * sizeof(bf16_t);      // 1.18 MB
  const size_t zbytes = (size_t)M_ * D_ * sizeof(bf16_t);      // 25.2 MB
  bf16_t* wq_bf = (bf16_t*)carve(wbytes);
  bf16_t* wk_bf = (bf16_t*)carve(wbytes);
  bf16_t* wv_bf = (bf16_t*)carve(wbytes);
  bf16_t* wo_bf = (bf16_t*)carve(wbytes);
  bf16_t* z_bf  = (bf16_t*)carve(zbytes);   // reused as ctx after QKV
  bf16_t* Qb    = (bf16_t*)carve(zbytes);   // [B,H,S,HD]
  bf16_t* Kb    = (bf16_t*)carve(zbytes);   // [B,H,S,HD]
  bf16_t* VTb   = (bf16_t*)carve(zbytes);   // [B,H,HD,S]
  bf16_t* ctx   = z_bf;                     // alias: z dead after QKV GEMMs

  // 1) fp32 -> bf16 conversions
  {
    int n4 = (M_ * D_) / 4;
    cvt_f32_bf16<<<n4 / 256, 256, 0, stream>>>((const float4*)z, z_bf, n4);
    int w4 = (D_ * D_) / 4;
    cvt_f32_bf16<<<w4 / 256, 256, 0, stream>>>((const float4*)W_q, wq_bf, w4);
    cvt_f32_bf16<<<w4 / 256, 256, 0, stream>>>((const float4*)W_k, wk_bf, w4);
    cvt_f32_bf16<<<w4 / 256, 256, 0, stream>>>((const float4*)W_v, wv_bf, w4);
    cvt_f32_bf16<<<w4 / 256, 256, 0, stream>>>((const float4*)W_o, wo_bf, w4);
  }

  // 2) QKV projections: (M/256)*(D/64) = 768 blocks of 8 waves (32 rows/wave)
  const int gemm_blocks = (M_ / 256) * (D_ / 64);
  gemm_qkv<<<gemm_blocks, 256, 0, stream>>>(z_bf, wq_bf, Qb, 0);
  gemm_qkv<<<gemm_blocks, 256, 0, stream>>>(z_bf, wk_bf, Kb, 0);
  gemm_qkv<<<gemm_blocks, 256, 0, stream>>>(z_bf, wv_bf, VTb, 1);

  // 3) flash attention: B*H*(S/64) = 3072 blocks of 4 waves
  flash_attn<<<B_ * H_ * (S_ / 64), 128, 0, stream>>>(Qb, Kb, VTb, ctx);

  // 4) output projection + bias (fp32 out)
  gemm_out<<<gemm_blocks, 256, 0, stream>>>(ctx, wo_bf, b_o, out);

  (void)in_sizes; (void)n_in; (void)out_size; (void)ws_size;
}
